// SAModule_68264210203013
// MI455X (gfx1250) — matrix operand order repaired and checked by reading, not compiled, not run on hardware
//
#include <hip/hip_runtime.h>

#define N_OBJ 2048
#define H_F   256
#define HEADS 4
#define DK    64

typedef __attribute__((ext_vector_type(16))) _Float16 v16h;
typedef __attribute__((ext_vector_type(8)))  _Float16 v8h;
typedef __attribute__((ext_vector_type(8)))  float    v8f;

static_assert(sizeof(_Float16) == 2, "f16 size");

#if defined(__has_builtin)
#if __has_builtin(__builtin_amdgcn_tensor_load_to_lds)
#define HAVE_TDM 1
#endif
#endif
#ifndef HAVE_TDM
#define HAVE_TDM 0
#endif

__device__ __forceinline__ v8f fzero8() {
  v8f z;
#pragma unroll
  for (int i = 0; i < 8; ++i) z[i] = 0.0f;
  return z;
}
__device__ __forceinline__ v16h hzero16() {
  v16h z;
#pragma unroll
  for (int i = 0; i < 16; ++i) z[i] = (_Float16)0.0f;
  return z;
}
__device__ __forceinline__ v16h cat16(v8h lo, v8h hi) {
  return __builtin_shufflevector(lo, hi, 0, 1, 2, 3, 4, 5, 6, 7, 8, 9, 10, 11, 12, 13, 14, 15);
}
__device__ __forceinline__ v8f wmma_f16(v16h a, v16h b, v8f c) {
  // D(16x16,f32) = A(16x32,f16) * B(32x16,f16) + C
  return __builtin_amdgcn_wmma_f32_16x16x32_f16(false, a, false, b, (short)0, c, false, false);
}

// A tile (16x32) from row-major f16 matrix: rows row0..+15, cols col0..+31.
// ISA 16-bit A layout: lane<16 -> row=lane, K {0..7,16..23}; lane>=16 -> row=lane-16, K {8..15,24..31}.
__device__ __forceinline__ v16h load_a16(const _Float16* p, int ld, int row0, int col0) {
  const int lane = threadIdx.x & 31;
  const int hf = lane >> 4;
  const _Float16* q = p + (size_t)(row0 + (lane & 15)) * ld + col0 + hf * 8;
  v8h lo = *(const v8h*)q;
  v8h hi = *(const v8h*)(q + 16);
  return cat16(lo, hi);
}

// B tile (32x16): B[k][n] = W[n][k], W row-major (out x in).
// ISA B layout: lane<16 -> col=lane, K {0..7,16..23}; lane>=16 -> col=lane-16, K {8..15,24..31}.
__device__ __forceinline__ v16h load_b16(const _Float16* p, int ld, int col0, int k0) {
  const int lane = threadIdx.x & 31;
  const int hf = lane >> 4;
  const _Float16* q = p + (size_t)(col0 + (lane & 15)) * ld + k0 + hf * 8;
  v8h lo = *(const v8h*)q;
  v8h hi = *(const v8h*)(q + 16);
  return cat16(lo, hi);
}

#if HAVE_TDM
typedef __attribute__((ext_vector_type(4))) unsigned int v4u;
typedef __attribute__((ext_vector_type(8))) int v8i;
typedef __attribute__((ext_vector_type(4))) int v4i;

__device__ __forceinline__ unsigned lds_offset_of(const void* p) {
  // flat LDS address: offset lives in the low 32 bits (ISA 10.2 aperture mapping)
  return (unsigned)(unsigned long long)p;
}

// 2-D tile DMA: global (f16, row stride `stride0` elems) -> LDS (packed tile_d0 x tile_d1).
// D# per CDNA5 ISA; groups 2/3 zero (<=2D tensor).
// This toolchain exposes the 6-arg builtin: (v4u g0, v8i g1, v4i, v4i, v8i, i32 cpol).
__device__ __forceinline__ void tdm_load_2d(const void* gptr, unsigned lds_off,
                                            unsigned tensor_d0, unsigned tensor_d1,
                                            unsigned tile_d0, unsigned tile_d1,
                                            unsigned stride0) {
  unsigned long long ga = (unsigned long long)gptr;
  unsigned ga_lo = (unsigned)__builtin_amdgcn_readfirstlane((int)(ga & 0xffffffffu));
  unsigned ga_hi = (unsigned)__builtin_amdgcn_readfirstlane((int)((ga >> 32) & 0x01ffffffu));
  unsigned lo    = (unsigned)__builtin_amdgcn_readfirstlane((int)lds_off);
  v4u g0;
  g0[0] = 1u;                        // count=1 user descriptor
  g0[1] = lo;                        // lds_addr (bytes)
  g0[2] = ga_lo;                     // global_addr[31:0]
  g0[3] = ga_hi | (2u << 30);        // global_addr[56:32] | type=2 ("image")
  v8i g1;
  g1[0] = (int)(1u << 16);           // data_size=1 (2 bytes); no multicast/pad/iterate
  g1[1] = (int)((tensor_d0 & 0xffffu) << 16);                      // tensor_dim0[15:0] -> bits[63:48]
  g1[2] = (int)((tensor_d0 >> 16) | ((tensor_d1 & 0xffffu) << 16));// tensor_dim0[31:16] | tensor_dim1[15:0]
  g1[3] = (int)((tensor_d1 >> 16) | (tile_d0 << 16));              // tensor_dim1[31:16] | tile_dim0
  g1[4] = (int)(tile_d1 & 0xffffu);                                // tile_dim1 (tile_dim2 = 0)
  g1[5] = (int)stride0;                                            // tensor_dim0_stride[31:0]
  g1[6] = 0;                                                       // stride0[47:32] | stride1[15:0]
  g1[7] = 0;
  v4i z4;
  z4[0] = 0; z4[1] = 0; z4[2] = 0; z4[3] = 0;
  v8i z8;
#pragma unroll
  for (int i = 0; i < 8; ++i) z8[i] = 0;
  __builtin_amdgcn_tensor_load_to_lds(g0, g1, z4, z4, z8, 0);
}
#endif

// ---------------------------------------------------------------- prep: f32 -> f16 (+K padding)
__global__ void k0_prep(const float* feat, const float* c1w, const float* c2w, const float* bbw,
                        const float* f1w, const float* f2w, const float* f3w,
                        const float* wq, const float* wk, const float* wv, const float* wo,
                        _Float16* featp, _Float16* c1p, _Float16* c2p, _Float16* bbp,
                        _Float16* f1p, _Float16* f2p, _Float16* f3p,
                        _Float16* wqp, _Float16* wkp, _Float16* wvp, _Float16* wop) {
  const int idx = blockIdx.x * blockDim.x + threadIdx.x;
  const int stride = gridDim.x * blockDim.x;
  for (int i = idx; i < N_OBJ * 256; i += stride) {  // obj_feature padded 240 -> 256
    int r = i >> 8, c = i & 255;
    featp[i] = (_Float16)(c < 240 ? feat[r * 240 + c] : 0.0f);
  }
  for (int i = idx; i < 256 * 256; i += stride) {    // conv1_w padded 240 -> 256
    int r = i >> 8, c = i & 255;
    c1p[i] = (_Float16)(c < 240 ? c1w[r * 240 + c] : 0.0f);
    c2p[i] = (_Float16)c2w[i];
    wqp[i] = (_Float16)wq[i];
    wkp[i] = (_Float16)wk[i];
    wvp[i] = (_Float16)wv[i];
    wop[i] = (_Float16)wo[i];
  }
  for (int i = idx; i < 256 * 32; i += stride) {     // bbox_w padded 6 -> 32
    int r = i >> 5, c = i & 31;
    bbp[i] = (_Float16)(c < 6 ? bbw[r * 6 + c] : 0.0f);
  }
  for (int i = idx; i < 128; i += stride) { f1p[i] = (_Float16)f1w[i]; f3p[i] = (_Float16)f3w[i]; }
  for (int i = idx; i < 1024; i += stride) { f2p[i] = (_Float16)f2w[i]; }
}

// ---------------------------------------------------------------- k1: h1 = PReLU(BN(feat @ W1^T + b1))
__global__ __launch_bounds__(32) void k1_feat(const _Float16* featp, const _Float16* c1p,
                                              const float* c1b, const float* bng, const float* bnb,
                                              const float* bnm, const float* bnv, const float* pa,
                                              _Float16* h1) {
  const int rt = blockIdx.x, ct = blockIdx.y;
  v8f acc = fzero8();
#pragma unroll
  for (int ks = 0; ks < 8; ++ks) {
    __builtin_prefetch(featp + (size_t)(rt * 16) * 256 + (ks + 1) * 32, 0, 3);
    v16h a = load_a16(featp, 256, rt * 16, ks * 32);
    v16h b = load_b16(c1p, 256, ct * 16, ks * 32);
    acc = wmma_f16(a, b, acc);
  }
  const int lane = threadIdx.x & 31, hf = lane >> 4, col = ct * 16 + (lane & 15);
  const float inv = rsqrtf(bnv[col] + 1e-5f);
  const float g = bng[col], bb = bnb[col], mu = bnm[col], slope = pa[col], cb = c1b[col];
#pragma unroll
  for (int r = 0; r < 8; ++r) {
    float y = acc[r] + cb;
    y = (y - mu) * inv * g + bb;
    y = y >= 0.0f ? y : slope * y;
    h1[(size_t)(rt * 16 + r + hf * 8) * H_F + col] = (_Float16)y;
  }
}

// ---------------------------------------------------------------- k2: x = h1@W2^T + b2 + [c,s]@Wb^T + bbb
__global__ __launch_bounds__(32) void k2_x(const _Float16* h1, const _Float16* c2p, const float* c2b,
                                           const float* center, const float* size_, const _Float16* bbp,
                                           const float* bbb, _Float16* x) {
  const int rt = blockIdx.x, ct = blockIdx.y;
  const int lane = threadIdx.x & 31, hf = lane >> 4, ln = lane & 15;
  v8f acc = fzero8();
#pragma unroll
  for (int ks = 0; ks < 8; ++ks) {
    v16h a = load_a16(h1, 256, rt * 16, ks * 32);
    v16h b = load_b16(c2p, 256, ct * 16, ks * 32);
    acc = wmma_f16(a, b, acc);
  }
  v16h ab = hzero16();
  if (hf == 0) {  // lanes>=16 cover K 8..15 / 24..31 which are zero padding
    int row = rt * 16 + ln;
    ab[0] = (_Float16)center[row * 3 + 0];
    ab[1] = (_Float16)center[row * 3 + 1];
    ab[2] = (_Float16)center[row * 3 + 2];
    ab[3] = (_Float16)size_[row * 3 + 0];
    ab[4] = (_Float16)size_[row * 3 + 1];
    ab[5] = (_Float16)size_[row * 3 + 2];
  }
  acc = wmma_f16(ab, load_b16(bbp, 32, ct * 16, 0), acc);
  const int col = ct * 16 + ln;
  const float bsum = c2b[col] + bbb[col];
#pragma unroll
  for (int r = 0; r < 8; ++r)
    x[(size_t)(rt * 16 + r + hf * 8) * H_F + col] = (_Float16)(acc[r] + bsum);
}

// ---------------------------------------------------------------- k3: Q,K as (h,n,dk) f16; V transposed (h,dk,n)
__global__ __launch_bounds__(32) void k3_qkv(const _Float16* x, const _Float16* wqp, const _Float16* wkp,
                                             const _Float16* wvp, const float* bq, const float* bk,
                                             const float* bv, _Float16* qo, _Float16* ko, _Float16* vto) {
  const int rt = blockIdx.x, ct = blockIdx.y, t = blockIdx.z;
  const _Float16* W = (t == 0) ? wqp : (t == 1) ? wkp : wvp;
  const float* bias = (t == 0) ? bq : (t == 1) ? bk : bv;
  v8f acc = fzero8();
#pragma unroll
  for (int ks = 0; ks < 8; ++ks) {
    v16h a = load_a16(x, 256, rt * 16, ks * 32);
    v16h b = load_b16(W, 256, ct * 16, ks * 32);
    acc = wmma_f16(a, b, acc);
  }
  const int lane = threadIdx.x & 31, hf = lane >> 4, col = ct * 16 + (lane & 15);
  const int head = col >> 6, d = col & 63;
  const float bc = bias[col];
#pragma unroll
  for (int r = 0; r < 8; ++r) {
    int row = rt * 16 + r + hf * 8;
    float y = acc[r] + bc;
    if (t == 0)      qo[(size_t)head * N_OBJ * DK + (size_t)row * DK + d] = (_Float16)y;
    else if (t == 1) ko[(size_t)head * N_OBJ * DK + (size_t)row * DK + d] = (_Float16)y;
    else             vto[(size_t)head * DK * N_OBJ + (size_t)d * N_OBJ + row] = (_Float16)y;
  }
}

// ---------------------------------------------------------------- k4: fused pair-MLP bias + flash attention
// block = 128 threads (4 waves); wave w == head w; blockIdx.x == 16-query tile.
// Per 32-key chunk: TDM DMAs the K and V^T tiles into LDS while the wave runs the
// pair-MLP WMMAs; s_wait_tensorcnt gates the score GEMMs.
__global__ __launch_bounds__(128) void k4_attn(const _Float16* qm, const _Float16* km, const _Float16* vt,
                                               const float* center,
                                               const _Float16* f1p, const float* f1b,
                                               const float* l1g, const float* l1b,
                                               const _Float16* f2p, const float* f2b,
                                               const float* l2g, const float* l2b,
                                               const _Float16* f3p, const float* f3b,
                                               _Float16* attn) {
  __shared__ __align__(16) float    bias_ld[512 * HEADS];   // pair-major (qi_local*32+kj_local)*4+head
  __shared__ __align__(16) float    hstage[HEADS][16 * 32]; // per-wave LN staging
  __shared__ __align__(16) _Float16 pstage[HEADS][16 * 32]; // per-wave P staging (A layout source)
#if HAVE_TDM
  __shared__ __align__(16) _Float16 kstage[HEADS][32 * DK]; // 32 keys x 64 dims
  __shared__ __align__(16) _Float16 vstage[HEADS][DK * 32]; // 64 dims x 32 keys
#endif

  const int wave = threadIdx.x >> 5;
  const int lane = threadIdx.x & 31;
  const int hf = lane >> 4;
  const int ln = lane & 15;
  const int q0 = blockIdx.x * 16;

  const _Float16* qh = qm + (size_t)wave * N_OBJ * DK;
  const v16h qa0 = load_a16(qh, DK, q0, 0);
  const v16h qa1 = load_a16(qh, DK, q0, 32);

  // Preload MLP B tiles (fc1 padded K 4->32, fc3 padded N 4->16)
  v16h b1[2], b2[2], b3;
#pragma unroll
  for (int nt = 0; nt < 2; ++nt) {
#pragma unroll
    for (int i = 0; i < 16; ++i) {
      b1[nt][i] = (hf == 0 && i < 4) ? f1p[(nt * 16 + ln) * 4 + i] : (_Float16)0.0f;
      b2[nt][i] = f2p[(nt * 16 + ln) * 32 + hf * 8 + i + (i >= 8 ? 8 : 0)];
    }
  }
#pragma unroll
  for (int i = 0; i < 16; ++i) b3[i] = (ln < 4) ? f3p[ln * 32 + hf * 8 + i + (i >= 8 ? 8 : 0)] : (_Float16)0.0f;

  float mrow[8], lrow[8];
  v8f o[4];
#pragma unroll
  for (int r = 0; r < 8; ++r) { mrow[r] = -1e30f; lrow[r] = 0.0f; }
#pragma unroll
  for (int nt = 0; nt < 4; ++nt) o[nt] = fzero8();

  float* hs = hstage[wave];
  _Float16* ps = pstage[wave];

  for (int ck = 0; ck < 64; ++ck) {
    const int k0 = ck * 32;

#if HAVE_TDM
    // Issue the K / V^T tile DMAs for this chunk; they run while the MLP computes.
    // (previous iteration ended with __syncthreads(), so prior ds reads of these
    //  buffers are complete before the TDM overwrites them)
    tdm_load_2d(km + (size_t)wave * N_OBJ * DK + (size_t)k0 * DK,
                lds_offset_of(&kstage[wave][0]),
                /*tensor_d0=*/DK, /*tensor_d1=*/32, /*tile_d0=*/DK, /*tile_d1=*/32,
                /*stride0=*/DK);
    tdm_load_2d(vt + (size_t)wave * DK * N_OBJ + k0,
                lds_offset_of(&vstage[wave][0]),
                /*tensor_d0=*/32, /*tensor_d1=*/DK, /*tile_d0=*/32, /*tile_d1=*/DK,
                /*stride0=*/N_OBJ);
#endif

    // ---- per-pair bias MLP: 512 pairs = 32 M-tiles, 8 per wave
    for (int mi = 0; mi < 8; ++mi) {
      const int mt = wave * 8 + mi;
      // A1: geometric features (dx,dy,dz,dist) in K 0..3, zero elsewhere
      v16h a1 = hzero16();
      if (hf == 0) {
        int p = mt * 16 + ln;
        int qi = q0 + (p >> 5), kj = k0 + (p & 31);
        float dx = center[kj * 3 + 0] - center[qi * 3 + 0];
        float dy = center[kj * 3 + 1] - center[qi * 3 + 1];
        float dz = center[kj * 3 + 2] - center[qi * 3 + 2];
        float dd = sqrtf(dx * dx + dy * dy + dz * dz);
        a1[0] = (_Float16)dx; a1[1] = (_Float16)dy; a1[2] = (_Float16)dz; a1[3] = (_Float16)dd;
      }
#pragma unroll
      for (int nt = 0; nt < 2; ++nt) {
        v8f c1 = wmma_f16(a1, b1[nt], fzero8());
#pragma unroll
        for (int r = 0; r < 8; ++r) {
          float y = c1[r] + f1b[nt * 16 + ln];
          hs[(r + hf * 8) * 32 + nt * 16 + ln] = y > 0.0f ? y : 0.0f;
        }
      }
      __syncthreads();
      if (hf == 0) {  // LayerNorm over 32 features, one row per lane
        float s = 0.0f, s2 = 0.0f;
#pragma unroll
        for (int c = 0; c < 32; ++c) { float v = hs[ln * 32 + c]; s += v; s2 += v * v; }
        float mean = s * (1.0f / 32.0f);
        float inv = rsqrtf(s2 * (1.0f / 32.0f) - mean * mean + 1e-5f);
#pragma unroll
        for (int c = 0; c < 32; ++c)
          hs[ln * 32 + c] = (hs[ln * 32 + c] - mean) * inv * l1g[c] + l1b[c];
      }
      __syncthreads();
      v16h a2;
#pragma unroll
      for (int i = 0; i < 8; ++i) {
        a2[i]     = (_Float16)hs[ln * 32 + hf * 8 + i];
        a2[8 + i] = (_Float16)hs[ln * 32 + 16 + hf * 8 + i];
      }
      __syncthreads();
#pragma unroll
      for (int nt = 0; nt < 2; ++nt) {
        v8f c2 = wmma_f16(a2, b2[nt], fzero8());
#pragma unroll
        for (int r = 0; r < 8; ++r) {
          float y = c2[r] + f2b[nt * 16 + ln];
          hs[(r + hf * 8) * 32 + nt * 16 + ln] = y > 0.0f ? y : 0.0f;
        }
      }
      __syncthreads();
      if (hf == 0) {
        float s = 0.0f, s2 = 0.0f;
#pragma unroll
        for (int c = 0; c < 32; ++c) { float v = hs[ln * 32 + c]; s += v; s2 += v * v; }
        float mean = s * (1.0f / 32.0f);
        float inv = rsqrtf(s2 * (1.0f / 32.0f) - mean * mean + 1e-5f);
#pragma unroll
        for (int c = 0; c < 32; ++c)
          hs[ln * 32 + c] = (hs[ln * 32 + c] - mean) * inv * l2g[c] + l2b[c];
      }
      __syncthreads();
      v16h a3;
#pragma unroll
      for (int i = 0; i < 8; ++i) {
        a3[i]     = (_Float16)hs[ln * 32 + hf * 8 + i];
        a3[8 + i] = (_Float16)hs[ln * 32 + 16 + hf * 8 + i];
      }
      v8f c3 = wmma_f16(a3, b3, fzero8());
      if (ln < 4) {
#pragma unroll
        for (int r = 0; r < 8; ++r)
          bias_ld[(mt * 16 + r + hf * 8) * HEADS + ln] = c3[r] + f3b[ln];
      }
      __syncthreads();
    }
    __syncthreads();  // bias_ld for this chunk complete (all waves)

    // ---- scores S = Q K^T / 8 + bias (two 16-key subtiles)
#if HAVE_TDM
    __builtin_amdgcn_s_wait_tensorcnt((unsigned short)0);  // K/V tiles landed in LDS
    const _Float16* kbuf = &kstage[wave][0];
#else
    const _Float16* kbuf = km + (size_t)wave * N_OBJ * DK;
#endif
    v8f st[2];
#pragma unroll
    for (int kt = 0; kt < 2; ++kt) {
#if HAVE_TDM
      v8f s = wmma_f16(qa0, load_b16(kbuf, DK, kt * 16, 0), fzero8());
      s = wmma_f16(qa1, load_b16(kbuf, DK, kt * 16, 32), s);
#else
      v8f s = wmma_f16(qa0, load_b16(kbuf, DK, k0 + kt * 16, 0), fzero8());
      s = wmma_f16(qa1, load_b16(kbuf, DK, k0 + kt * 16, 32), s);
#endif
#pragma unroll
      for (int r = 0; r < 8; ++r)
        s[r] = s[r] * 0.125f + bias_ld[((r + hf * 8) * 32 + kt * 16 + ln) * HEADS + wave];
      st[kt] = s;
    }

    // ---- online softmax (row reductions within 16-lane halves match the C layout)
#pragma unroll
    for (int r = 0; r < 8; ++r) {
      float mx = fmaxf(st[0][r], st[1][r]);
#pragma unroll
      for (int m = 1; m <= 8; m <<= 1) mx = fmaxf(mx, __shfl_xor(mx, m, 32));
      float mn = fmaxf(mrow[r], mx);
      float sc = __expf(mrow[r] - mn);
      float e0 = __expf(st[0][r] - mn);
      float e1 = __expf(st[1][r] - mn);
      float rs = e0 + e1;
#pragma unroll
      for (int m = 1; m <= 8; m <<= 1) rs += __shfl_xor(rs, m, 32);
      lrow[r] = lrow[r] * sc + rs;
      mrow[r] = mn;
#pragma unroll
      for (int nt = 0; nt < 4; ++nt) o[nt][r] *= sc;
      ps[(r + hf * 8) * 32 + ln] = (_Float16)e0;
      ps[(r + hf * 8) * 32 + 16 + ln] = (_Float16)e1;
    }
    __syncthreads();  // ps visible (and bias_ld reads done before next chunk's writes)

    // ---- O += P V (P 16x32 from LDS in A layout; V^T gives contiguous B tiles)
    v16h pa = load_a16(ps, 32, 0, 0);
#pragma unroll
    for (int nt = 0; nt < 4; ++nt) {
#if HAVE_TDM
      o[nt] = wmma_f16(pa, load_b16(&vstage[wave][0], 32, nt * 16, 0), o[nt]);
#else
      o[nt] = wmma_f16(pa, load_b16(vt + (size_t)wave * DK * N_OBJ, N_OBJ, nt * 16, k0), o[nt]);
#endif
    }
    __syncthreads();
  }

  // ---- epilogue: normalize and store per-head output into (N, 256) f16
#pragma unroll
  for (int nt = 0; nt < 4; ++nt) {
#pragma unroll
    for (int r = 0; r < 8; ++r) {
      float y = o[nt][r] / lrow[r];
      attn[(size_t)(q0 + r + hf * 8) * H_F + wave * 64 + nt * 16 + ln] = (_Float16)y;
    }
  }
}

// ---------------------------------------------------------------- k5: out = attn @ Wo^T + bo (f32 out)
__global__ __launch_bounds__(32) void k5_out(const _Float16* attn, const _Float16* wop,
                                             const float* bo, float* out) {
  const int rt = blockIdx.x, ct = blockIdx.y;
  v8f acc = fzero8();
#pragma unroll
  for (int ks = 0; ks < 8; ++ks) {
    v16h a = load_a16(attn, 256, rt * 16, ks * 32);
    v16h b = load_b16(wop, 256, ct * 16, ks * 32);
    acc = wmma_f16(a, b, acc);
  }
  const int lane = threadIdx.x & 31, hf = lane >> 4, col = ct * 16 + (lane & 15);
  const float bc = bo[col];
#pragma unroll
  for (int r = 0; r < 8; ++r)
    out[(size_t)(rt * 16 + r + hf * 8) * H_F + col] = acc[r] + bc;
}

// ---------------------------------------------------------------- launch
extern "C" void kernel_launch(void* const* d_in, const int* in_sizes, int n_in,
                              void* d_out, int out_size, void* d_ws, size_t ws_size,
                              hipStream_t stream) {
  const float* obj_feature = (const float*)d_in[0];
  const float* obj_center  = (const float*)d_in[1];
  const float* obj_size    = (const float*)d_in[2];
  const float* conv1_w = (const float*)d_in[3];
  const float* conv1_b = (const float*)d_in[4];
  const float* bn_g = (const float*)d_in[5];
  const float* bn_b = (const float*)d_in[6];
  const float* bn_mean = (const float*)d_in[7];
  const float* bn_var = (const float*)d_in[8];
  const float* prelu_a = (const float*)d_in[9];
  const float* conv2_w = (const float*)d_in[10];
  const float* conv2_b = (const float*)d_in[11];
  const float* bbox_w = (const float*)d_in[12];
  const float* bbox_b = (const float*)d_in[13];
  const float* fc1_w = (const float*)d_in[14];
  const float* fc1_b = (const float*)d_in[15];
  const float* ln1_g = (const float*)d_in[16];
  const float* ln1_b = (const float*)d_in[17];
  const float* fc2_w = (const float*)d_in[18];
  const float* fc2_b = (const float*)d_in[19];
  const float* ln2_g = (const float*)d_in[20];
  const float* ln2_b = (const float*)d_in[21];
  const float* fc3_w = (const float*)d_in[22];
  const float* fc3_b = (const float*)d_in[23];
  const float* wq = (const float*)d_in[24];
  const float* bq = (const float*)d_in[25];
  const float* wk = (const float*)d_in[26];
  const float* bk = (const float*)d_in[27];
  const float* wv = (const float*)d_in[28];
  const float* bv = (const float*)d_in[29];
  const float* wo = (const float*)d_in[30];
  const float* bo = (const float*)d_in[31];

  _Float16* ws = (_Float16*)d_ws;
  // element offsets (all multiples of 8 halves -> 16B aligned)
  size_t off = 0;
  _Float16* featp = ws + off; off += (size_t)N_OBJ * 256;
  _Float16* c1p = ws + off; off += 256 * 256;
  _Float16* c2p = ws + off; off += 256 * 256;
  _Float16* bbp = ws + off; off += 256 * 32;
  _Float16* f1p = ws + off; off += 128;
  _Float16* f2p = ws + off; off += 1024;
  _Float16* f3p = ws + off; off += 128;
  _Float16* wqp = ws + off; off += 256 * 256;
  _Float16* wkp = ws + off; off += 256 * 256;
  _Float16* wvp = ws + off; off += 256 * 256;
  _Float16* wop = ws + off; off += 256 * 256;
  _Float16* h1 = ws + off; off += (size_t)N_OBJ * 256;
  _Float16* x  = ws + off; off += (size_t)N_OBJ * 256;
  _Float16* qm = ws + off; off += (size_t)N_OBJ * 256;
  _Float16* km = ws + off; off += (size_t)N_OBJ * 256;
  _Float16* vt = ws + off; off += (size_t)N_OBJ * 256;
  _Float16* attn = ws + off; off += (size_t)N_OBJ * 256;

  k0_prep<<<512, 256, 0, stream>>>(obj_feature, conv1_w, conv2_w, bbox_w, fc1_w, fc2_w, fc3_w,
                                   wq, wk, wv, wo,
                                   featp, c1p, c2p, bbp, f1p, f2p, f3p, wqp, wkp, wvp, wop);
  k1_feat<<<dim3(N_OBJ / 16, 16), 32, 0, stream>>>(featp, c1p, conv1_b, bn_g, bn_b, bn_mean, bn_var,
                                                   prelu_a, h1);
  k2_x<<<dim3(N_OBJ / 16, 16), 32, 0, stream>>>(h1, c2p, conv2_b, obj_center, obj_size, bbp, bbox_b, x);
  k3_qkv<<<dim3(N_OBJ / 16, 16, 3), 32, 0, stream>>>(x, wqp, wkp, wvp, bq, bk, bv, qm, km, vt);
  k4_attn<<<N_OBJ / 16, 128, 0, stream>>>(qm, km, vt, obj_center, f1p, fc1_b, ln1_g, ln1_b,
                                          f2p, fc2_b, ln2_g, ln2_b, f3p, fc3_b, attn);
  k5_out<<<dim3(N_OBJ / 16, 16), 32, 0, stream>>>(attn, wop, bo, (float*)d_out);
}
